// LOCPod_15839839387907
// MI455X (gfx1250) — matrix operand order repaired and checked by reading, not compiled, not run on hardware
//
#include <hip/hip_runtime.h>
#include <hip/hip_bf16.h>

typedef __bf16 bf16;
typedef __attribute__((ext_vector_type(16))) __bf16 v16b;
typedef __attribute__((ext_vector_type(8)))  float  v8f;

#define GAIN 2.5f
#define EPSF 1e-6f

union V16B { v16b v; uint4 q[2]; };
union Pack8 { uint4 q; bf16 e[8]; };

__device__ inline bf16 f2bf(float f) {
    unsigned u = __builtin_bit_cast(unsigned, f);
    unsigned r = (u + 0x7FFFu + ((u >> 16) & 1u)) >> 16;
    return __builtin_bit_cast(bf16, (unsigned short)r);
}

__device__ inline v8f zero8() {
    v8f z;
    #pragma unroll
    for (int i = 0; i < 8; ++i) z[i] = 0.f;
    return z;
}

__device__ inline v8f wmma_bf(v16b a, v16b b, v8f c) {
    return __builtin_amdgcn_wmma_f32_16x16x32_bf16(false, a, false, b, (short)0, c, false, false);
}

// A-tile (16x32 bf16) from row-major source: lane m = lane&15,
// low 8 elems at byte (lane>>4)*16, high 8 elems at +32 bytes. (ISA 7.12.2)
__device__ inline v16b loadA(const char* base, int rowStrideB, int lane) {
    const char* p = base + (size_t)(lane & 15) * rowStrideB + ((lane >> 4) << 4);
    V16B r;
    r.q[0] = *(const uint4*)(p);
    r.q[1] = *(const uint4*)(p + 32);
    return r.v;
}
// B-tile (32x16 bf16) from transposed storage Bt[n][k] row-major:
// lane col n = lane&15, low 8 k at k-offset (lane>>4)*8 elems, high 8 k at +16 elems.
__device__ inline v16b loadB(const char* base, int rowStrideB, int lane) {
    const char* p = base + (size_t)(lane & 15) * rowStrideB + ((lane >> 4) << 4);
    V16B r;
    r.q[0] = *(const uint4*)(p);
    r.q[1] = *(const uint4*)(p + 32);
    return r.v;
}

__device__ inline float gelu_exact(float x) {
    return 0.5f * x * (1.f + erff(x * 0.70710678118654752f));
}

// ---------------------------------------------------------------------------
// 1) Per-u modulated weights, stored transposed bf16: WT[(wid*8+u)][out][in]
//    WT'[o][c] = (codes[u] @ Wc)[c] * W[c][o]
// ---------------------------------------------------------------------------
__global__ __launch_bounds__(256) void precompute_wt(
    const float* __restrict__ codes,
    const float* __restrict__ qWc, const float* __restrict__ qW,
    const float* __restrict__ kWc, const float* __restrict__ kW,
    const float* __restrict__ vWc, const float* __restrict__ vW,
    const float* __restrict__ m1Wc, const float* __restrict__ m1W,
    const float* __restrict__ m2Wc, const float* __restrict__ m2W,
    bf16* __restrict__ WT)
{
    __shared__ float s[256];
    int wid = blockIdx.x, u = blockIdx.y, t = threadIdx.x;
    const float* Wc; const float* W;
    switch (wid) {
        case 0:  Wc = qWc;  W = qW;  break;
        case 1:  Wc = kWc;  W = kW;  break;
        case 2:  Wc = vWc;  W = vW;  break;
        case 3:  Wc = m1Wc; W = m1W; break;
        default: Wc = m2Wc; W = m2W; break;
    }
    const float* cu = codes + u * 192;
    float acc = 0.f;
    for (int j = 0; j < 192; ++j) acc += cu[j] * Wc[j * 256 + t];
    s[t] = acc;
    __syncthreads();
    bf16* out = WT + ((size_t)(wid * 8 + u) * 256 + t) * 256;
    for (int c = 0; c < 256; ++c) out[c] = f2bf(s[c] * W[c * 256 + t]);
}

// ---------------------------------------------------------------------------
// 2) LayerNorm over C=256, one wave32 per token, bf16 out
// ---------------------------------------------------------------------------
__global__ __launch_bounds__(256) void ln_kernel(
    const float* __restrict__ x, const float* __restrict__ g,
    const float* __restrict__ b, bf16* __restrict__ out)
{
    int wv = threadIdx.x >> 5, lane = threadIdx.x & 31;
    int token = blockIdx.x * 8 + wv;
    const float* xr = x + (size_t)token * 256 + lane * 8;
    float v[8];
    #pragma unroll
    for (int i = 0; i < 8; ++i) v[i] = xr[i];
    float s = 0.f;
    #pragma unroll
    for (int i = 0; i < 8; ++i) s += v[i];
    #pragma unroll
    for (int m = 16; m > 0; m >>= 1) s += __shfl_xor(s, m, 32);
    float mean = s * (1.f / 256.f);
    float q = 0.f;
    #pragma unroll
    for (int i = 0; i < 8; ++i) { float d = v[i] - mean; q += d * d; }
    #pragma unroll
    for (int m = 16; m > 0; m >>= 1) q += __shfl_xor(q, m, 32);
    float rstd = rsqrtf(q * (1.f / 256.f) + 1e-5f);
    Pack8 p;
    #pragma unroll
    for (int i = 0; i < 8; ++i) {
        int ch = lane * 8 + i;
        p.e[i] = f2bf((v[i] - mean) * rstd * g[ch] + b[ch]);
    }
    *(uint4*)(out + (size_t)token * 256 + lane * 8) = p.q;
}

// ---------------------------------------------------------------------------
// 3) Generic 1024x256 @ 256x256 WMMA GEMM, one 16x16 tile per wave.
//    modes: 0 store bf16 | 1 +bias, store V transposed [h][kq][tok]
//           2 +bias,gelu bf16 | 3 +bias,gelu, final gated blend -> f32 out
// ---------------------------------------------------------------------------
__global__ __launch_bounds__(256) void gemm256(
    const bf16* __restrict__ X,   // [BU][1024][256]
    const bf16* __restrict__ WT,  // [U][256(out)][256(in)]
    const float* __restrict__ bias, int mode,
    bf16* __restrict__ outB, bf16* __restrict__ outVt,
    const float* __restrict__ attn_in, const float* __restrict__ aff,
    float* __restrict__ outF)
{
    int bu = blockIdx.y, u = bu & 7;
    int wave = threadIdx.x >> 5, lane = threadIdx.x & 31;
    int t = blockIdx.x * 8 + wave;            // 0..1023 tiles
    int r0 = (t >> 4) * 16, c0 = (t & 15) * 16;
    const char* Xb = (const char*)(X + ((size_t)bu * 1024 + r0) * 256);
    const char* Wb = (const char*)(WT + ((size_t)u * 256 + c0) * 256);
    v8f acc = zero8();
    #pragma unroll
    for (int k = 0; k < 256; k += 32) {
        v16b a = loadA(Xb + k * 2, 512, lane);
        v16b b = loadB(Wb + k * 2, 512, lane);
        acc = wmma_bf(a, b, acc);
    }
    int n = lane & 15, half = lane >> 4;
    int col = c0 + n;
    float bv = bias ? bias[col] : 0.f;

    if (mode == 0) {
        bf16* dst = outB + ((size_t)bu * 1024 + r0 + half * 8) * 256 + col;
        #pragma unroll
        for (int r = 0; r < 8; ++r) dst[(size_t)r * 256] = f2bf(acc[r]);
    } else if (mode == 1) {
        Pack8 p;
        #pragma unroll
        for (int r = 0; r < 8; ++r) p.e[r] = f2bf(acc[r] + bv);
        int h = col >> 6, kq = col & 63;
        bf16* dst = outVt + (((size_t)bu * 4 + h) * 64 + kq) * 1024 + r0 + half * 8;
        *(uint4*)dst = p.q;
    } else if (mode == 2) {
        bf16* dst = outB + ((size_t)bu * 1024 + r0 + half * 8) * 256 + col;
        #pragma unroll
        for (int r = 0; r < 8; ++r) dst[(size_t)r * 256] = f2bf(gelu_exact(acc[r] + bv));
    } else {
        #pragma unroll
        for (int r = 0; r < 8; ++r) {
            int row = r0 + half * 8 + r;
            size_t idx = ((size_t)bu * 1024 + row) * 256 + col;
            float gt = aff[(size_t)bu * 1024 + row];
            float z  = gelu_exact(acc[r] + bv);
            outF[idx] = (1.f - gt) * attn_in[idx] + gt * GAIN * (z * gt);
        }
    }
}

// ---------------------------------------------------------------------------
// 4) Flash attention with affinity renormalization.
//    grid (V/64, B*U*H), block 128 (4 waves, 16 q-rows each).
// ---------------------------------------------------------------------------
__global__ __launch_bounds__(128) void attn_kernel(
    const bf16* __restrict__ Qb, const bf16* __restrict__ Kb,
    const bf16* __restrict__ Vt, const float* __restrict__ aff,
    const float* __restrict__ vars, float* __restrict__ attn_out)
{
    __shared__ __align__(16) float sS[4][16][33];
    __shared__ __align__(16) bf16  sP[4][16][32];
    __shared__ float sScale[4][16];
    __shared__ float sAff[4][32];
    __shared__ float sW[4][16], sL[4][16];

    int wv = threadIdx.x >> 5, lane = threadIdx.x & 31;
    int gy = blockIdx.y, h = gy & 3, bu = gy >> 2;
    int q0 = blockIdx.x * 64 + wv * 16;
    int n = lane & 15, half = lane >> 4;

    const char* Qbase = (const char*)(Qb + ((size_t)bu * 1024 + q0) * 256 + h * 64);
    const char* Kbase = (const char*)(Kb + (size_t)bu * 1024 * 256 + h * 64);
    const char* Vbase = (const char*)(Vt + ((size_t)bu * 4 + h) * 64 * 1024);
    const float* affR = aff + (size_t)bu * 1024;

    v16b aqLo = loadA(Qbase, 512, lane);        // kq 0..31
    v16b aqHi = loadA(Qbase + 64, 512, lane);   // kq 32..63

    v8f acc0 = zero8(), acc1 = zero8(), acc2 = zero8(), acc3 = zero8();
    float mrow = -1e30f, lrow = 0.f, wrow = 0.f;  // valid in lanes 0..15

    for (int kb = 0; kb < 1024; kb += 32) {
        if (kb + 32 < 1024)
            __builtin_prefetch(Kbase + (size_t)(kb + 32) * 512, 0, 1);
        // ----- scores: two 16-key sub-tiles -----
        #pragma unroll
        for (int sub = 0; sub < 2; ++sub) {
            const char* Ks = Kbase + (size_t)(kb + sub * 16) * 512;
            v8f s = zero8();
            s = wmma_bf(aqLo, loadB(Ks,      512, lane), s);
            s = wmma_bf(aqHi, loadB(Ks + 64, 512, lane), s);
            #pragma unroll
            for (int r = 0; r < 8; ++r) sS[wv][r + half * 8][sub * 16 + n] = s[r];
        }
        sAff[wv][lane] = affR[kb + lane];
        // ----- online softmax row pass (lanes 0..15, row = lane) -----
        if (lane < 16) {
            float cmax = -1e30f;
            for (int j = 0; j < 32; ++j) cmax = fmaxf(cmax, sS[wv][lane][j]);
            float mn = fmaxf(mrow, cmax);
            float sc = __expf(mrow - mn);
            float pl = 0.f, pw = 0.f;
            for (int j = 0; j < 32; ++j) {
                float p = __expf(sS[wv][lane][j] - mn);
                float a = sAff[wv][j];
                pl += p; pw += a * p;
                sP[wv][lane][j] = f2bf(a * p);
            }
            lrow = lrow * sc + pl;
            wrow = wrow * sc + pw;
            mrow = mn;
            sScale[wv][lane] = sc;
        }
        // ----- rescale accumulators (per-row scale) -----
        float scv[8];
        #pragma unroll
        for (int r = 0; r < 8; ++r) scv[r] = sScale[wv][r + half * 8];
        #pragma unroll
        for (int r = 0; r < 8; ++r) {
            acc0[r] *= scv[r]; acc1[r] *= scv[r];
            acc2[r] *= scv[r]; acc3[r] *= scv[r];
        }
        // ----- P (16x32) @ Vt chunk -> 4 col-groups of 16 -----
        V16B ap;
        {
            int off = half * 8;  // bf16 elements
            ap.q[0] = *(const uint4*)&sP[wv][n][off];
            ap.q[1] = *(const uint4*)&sP[wv][n][off + 16];
        }
        const char* Vs = Vbase + (size_t)kb * 2;
        acc0 = wmma_bf(ap.v, loadB(Vs,             2048, lane), acc0);
        acc1 = wmma_bf(ap.v, loadB(Vs + 16 * 2048, 2048, lane), acc1);
        acc2 = wmma_bf(ap.v, loadB(Vs + 32 * 2048, 2048, lane), acc2);
        acc3 = wmma_bf(ap.v, loadB(Vs + 48 * 2048, 2048, lane), acc3);
    }
    if (lane < 16) { sW[wv][lane] = wrow; sL[wv][lane] = lrow; }
    // ----- epilogue: renorm + gated residual, f32 out -----
    #pragma unroll
    for (int r = 0; r < 8; ++r) {
        int row = q0 + half * 8 + r;
        float a  = affR[row];
        float wr = sW[wv][r + half * 8];
        float lr = sL[wv][r + half * 8];
        float inv = 1.f / (a * wr + EPSF * lr);
        size_t base = ((size_t)bu * 1024 + row) * 256 + h * 64;
        float o0 = a * acc0[r] * inv, o1 = a * acc1[r] * inv;
        float o2 = a * acc2[r] * inv, o3 = a * acc3[r] * inv;
        attn_out[base +  0 + n] = (1.f - a) * vars[base +  0 + n] + a * GAIN * o0;
        attn_out[base + 16 + n] = (1.f - a) * vars[base + 16 + n] + a * GAIN * o1;
        attn_out[base + 32 + n] = (1.f - a) * vars[base + 32 + n] + a * GAIN * o2;
        attn_out[base + 48 + n] = (1.f - a) * vars[base + 48 + n] + a * GAIN * o3;
    }
}

// ---------------------------------------------------------------------------
extern "C" void kernel_launch(void* const* d_in, const int* in_sizes, int n_in,
                              void* d_out, int out_size, void* d_ws, size_t ws_size,
                              hipStream_t stream) {
    (void)in_sizes; (void)n_in; (void)out_size; (void)ws_size;
    const float* vars  = (const float*)d_in[0];
    const float* codes = (const float*)d_in[1];
    const float* affin = (const float*)d_in[2];
    const float* ln1g  = (const float*)d_in[3];
    const float* ln1b  = (const float*)d_in[4];
    const float* ln2g  = (const float*)d_in[5];
    const float* ln2b  = (const float*)d_in[6];
    const float* qWc   = (const float*)d_in[7];
    const float* qW    = (const float*)d_in[8];
    const float* kWc   = (const float*)d_in[9];
    const float* kW    = (const float*)d_in[10];
    const float* vWc   = (const float*)d_in[11];
    const float* vW    = (const float*)d_in[12];
    const float* vB    = (const float*)d_in[13];
    const float* m1Wc  = (const float*)d_in[14];
    const float* m1W   = (const float*)d_in[15];
    const float* m1B   = (const float*)d_in[16];
    const float* m2Wc  = (const float*)d_in[17];
    const float* m2W   = (const float*)d_in[18];
    const float* m2B   = (const float*)d_in[19];
    float* results = (float*)d_out;

    char* ws = (char*)d_ws;
    const size_t ACT_BF = (size_t)16384 * 256 * 2;   // 8.39 MB
    bf16* WT  = (bf16*)(ws);                          // 5*8*256*256*2 = 5.24 MB
    bf16* nv  = (bf16*)(ws + 5242880);
    bf16* Qb  = (bf16*)(ws + 5242880 + ACT_BF);
    bf16* Kb  = (bf16*)(ws + 5242880 + 2 * ACT_BF);
    bf16* Vt  = (bf16*)(ws + 5242880 + 3 * ACT_BF);
    float* ao = (float*)(ws + 5242880 + 4 * ACT_BF);  // 16.78 MB f32
    bf16* x2  = (bf16*)(ws + 5242880 + 4 * ACT_BF + (size_t)16384 * 256 * 4);
    bf16* h1  = (bf16*)((char*)x2 + ACT_BF);

    const size_t WSTEP = (size_t)8 * 256 * 256;       // elements per weight group

    precompute_wt<<<dim3(5, 8), 256, 0, stream>>>(
        codes, qWc, qW, kWc, kW, vWc, vW, m1Wc, m1W, m2Wc, m2W, WT);
    ln_kernel<<<2048, 256, 0, stream>>>(vars, ln1g, ln1b, nv);
    gemm256<<<dim3(128, 16), 256, 0, stream>>>(nv, WT + 0 * WSTEP, nullptr, 0,
        Qb, nullptr, nullptr, nullptr, nullptr);
    gemm256<<<dim3(128, 16), 256, 0, stream>>>(nv, WT + 1 * WSTEP, nullptr, 0,
        Kb, nullptr, nullptr, nullptr, nullptr);
    gemm256<<<dim3(128, 16), 256, 0, stream>>>(nv, WT + 2 * WSTEP, vB, 1,
        nullptr, Vt, nullptr, nullptr, nullptr);
    attn_kernel<<<dim3(16, 64), 128, 0, stream>>>(Qb, Kb, Vt, affin, vars, ao);
    ln_kernel<<<2048, 256, 0, stream>>>(ao, ln2g, ln2b, x2);
    gemm256<<<dim3(128, 16), 256, 0, stream>>>(x2, WT + 3 * WSTEP, m1B, 2,
        h1, nullptr, nullptr, nullptr, nullptr);
    gemm256<<<dim3(128, 16), 256, 0, stream>>>(h1, WT + 4 * WSTEP, m2B, 3,
        nullptr, nullptr, ao, affin, results);
}
